// VQ_CsiNet_69509750718556
// MI455X (gfx1250) — matrix operand order repaired and checked by reading, not compiled, not run on hardware
//
#include <hip/hip_runtime.h>

// ---------------------------------------------------------------------------
// CDNA5 / gfx1250 WMMA + TDM types
// ---------------------------------------------------------------------------
typedef __attribute__((ext_vector_type(16))) __bf16 v16bf;
typedef __attribute__((ext_vector_type(8)))  float  v8f;
typedef unsigned int u32x4 __attribute__((ext_vector_type(4)));
typedef int          i32x8 __attribute__((ext_vector_type(8)));
typedef int          i32x4 __attribute__((ext_vector_type(4)));

struct __align__(16) Q16 { unsigned int w[4]; };
union FragAB { v16bf v; Q16 q[2]; };

#if defined(__has_builtin)
#if __has_builtin(__builtin_amdgcn_tensor_load_to_lds) && __has_builtin(__builtin_amdgcn_s_wait_tensorcnt)
#define HAVE_TDM 1
#endif
#endif
#ifndef HAVE_TDM
#define HAVE_TDM 0
#endif

__device__ __forceinline__ float bf2f(unsigned short s) {
    unsigned int u = ((unsigned int)s) << 16;
    float f; __builtin_memcpy(&f, &u, 4); return f;
}
__device__ __forceinline__ unsigned short f2bf(float f) {
    unsigned int u; __builtin_memcpy(&u, &f, 4);
    u += 0x7FFFu + ((u >> 16) & 1u);            // round-to-nearest-even
    return (unsigned short)(u >> 16);
}
__device__ __forceinline__ float lrelu(float v) { return v >= 0.f ? v : 0.3f * v; }

// ---------------------------------------------------------------------------
// TDM: async-load a 2D bf16 tile [nRows x rowLen] (row stride = rowLen elems)
// from global memory into LDS at byte offset ldsOff.  Issued by one wave.
// Descriptor layout per CDNA5 ISA ch.8 (D# groups 0/1; 2D => groups 2,3 zero).
// This toolchain's builtin takes the 6-arg (clang-23 / therock-10.0) form:
//   (u32x4 g0, i32x8 g1, i32x4 g2, i32x4 g3, i32x8 extra, i32 cpol)
// ---------------------------------------------------------------------------
__device__ __forceinline__ void tdm_load_2d_bf16(const void* gptr, unsigned ldsOff,
                                                 unsigned rowLen, unsigned nRows,
                                                 unsigned tensorRows) {
#if HAVE_TDM
    unsigned long long ga = (unsigned long long)gptr;
    u32x4 g0;
    g0[0] = 1u;                                            // count=1, user desc
    g0[1] = ldsOff;                                        // lds_addr (bytes)
    g0[2] = (unsigned)(ga & 0xffffffffu);                  // global_addr[31:0]
    g0[3] = (unsigned)((ga >> 32) & 0x01ffffffu)           // global_addr[56:32]
          | (2u << 30);                                    // type=2 ("image")
    i32x8 g1;
    g1[0] = (int)(1u << 16);                               // data_size=1 -> 2 bytes
    g1[1] = (int)((rowLen & 0xffffu) << 16);               // tensor_dim0[15:0]
    g1[2] = (int)(((rowLen >> 16) & 0xffffu)
          | ((tensorRows & 0xffffu) << 16));               // dim0[31:16] | dim1[15:0]
    g1[3] = (int)(((tensorRows >> 16) & 0xffffu)
          | ((rowLen & 0xffffu) << 16));                   // dim1[31:16] | tile_dim0
    g1[4] = (int)(nRows & 0xffffu);                        // tile_dim1 (tile_dim2=0)
    g1[5] = (int)rowLen;                                   // tensor_dim0_stride[31:0]
    g1[6] = 0;
    g1[7] = 0;
    i32x4 z4 = {0, 0, 0, 0};
    i32x8 z8 = {0, 0, 0, 0, 0, 0, 0, 0};
    __builtin_amdgcn_tensor_load_to_lds(g0, g1, z4, z4, z8, 0);
    __builtin_amdgcn_s_wait_tensorcnt((short)0);
#else
    (void)gptr; (void)ldsOff; (void)rowLen; (void)nRows; (void)tensorRows;
#endif
}

// ---------------------------------------------------------------------------
// small utility kernels
// ---------------------------------------------------------------------------
__global__ void zero_f32(float* p, int n) {
    int i = threadIdx.x;
    if (i < n) p[i] = 0.f;
}

__global__ void cvt_bf16_kernel(const float* __restrict__ in,
                                unsigned short* __restrict__ out, int n) {
    int i = blockIdx.x * blockDim.x + threadIdx.x;
    if (i < n) out[i] = f2bf(in[i]);
}

__global__ void cnorm_kernel(const float* __restrict__ cb, float* __restrict__ cn) {
    int n = blockIdx.x * blockDim.x + threadIdx.x;
    if (n < 2048) {
        const float* r = cb + (size_t)n * 32;
        float s = 0.f;
        #pragma unroll
        for (int e = 0; e < 32; ++e) s += r[e] * r[e];
        cn[n] = s;
    }
}

__global__ void bn_finalize(const float* __restrict__ ssum, const float* __restrict__ ssq,
                            const float* __restrict__ g, const float* __restrict__ b,
                            float* __restrict__ scale, float* __restrict__ shift,
                            int C, float invN) {
    int c = threadIdx.x;
    if (c < C) {
        float mu  = ssum[c] * invN;
        float var = ssq[c] * invN - mu * mu;
        float sc  = g[c] * rsqrtf(var + 1e-5f);
        scale[c] = sc;
        shift[c] = b[c] - mu * sc;
    }
}

__global__ void bnlrelu_kernel(const unsigned short* __restrict__ t,
                               const float* __restrict__ scale,
                               const float* __restrict__ shift,
                               unsigned short* __restrict__ yb) {
    size_t i = (size_t)blockIdx.x * blockDim.x + threadIdx.x;
    int c = (int)((i >> 10) & 1);
    yb[i] = f2bf(lrelu(bf2f(t[i]) * scale[c] + shift[c]));
}

__global__ void residual_kernel(float* __restrict__ y,
                                const unsigned short* __restrict__ r,
                                const float* __restrict__ scale,
                                const float* __restrict__ shift) {
    size_t i = (size_t)blockIdx.x * blockDim.x + threadIdx.x;
    int c = (int)((i >> 10) & 1);
    float rv = bf2f(r[i]) * scale[c] + shift[c];
    y[i] = lrelu(y[i] + rv);
}

__global__ void sigmoid_kernel(const unsigned short* __restrict__ f,
                               const float* __restrict__ scale,
                               const float* __restrict__ shift,
                               float* __restrict__ out) {
    size_t i = (size_t)blockIdx.x * blockDim.x + threadIdx.x;
    int c = (int)((i >> 10) & 1);
    float v = bf2f(f[i]) * scale[c] + shift[c];
    out[i] = 1.f / (1.f + __expf(-v));
}

__global__ void loss_write_kernel(const float* __restrict__ acc, float* __restrict__ out) {
    // vq_loss = q_loss + 0.25*e_loss = 1.25 * mean((q - z)^2), mean over 65536*32
    out[0] = 1.25f * acc[0] * (1.0f / 2097152.0f);
}

// ---------------------------------------------------------------------------
// direct 3x3 SAME conv, NCHW, 32x32 spatial, raw bf16 output + batch stats.
// ---------------------------------------------------------------------------
template<int IC, int OC, bool BF16IN>
__global__ void conv3x3_kernel(const float* __restrict__ inF,
                               const unsigned short* __restrict__ inB,
                               const float* __restrict__ inScale,
                               const float* __restrict__ inShift,
                               const float* __restrict__ wgt,
                               const float* __restrict__ bias,
                               unsigned short* __restrict__ outp,
                               float* __restrict__ gsum,
                               float* __restrict__ gsq) {
    __shared__ float sw[OC * IC * 9];
    __shared__ float ssum[OC], ssq[OC];
    const int t = threadIdx.x;
    for (int i = t; i < OC * IC * 9; i += blockDim.x) sw[i] = wgt[i];
    if (t < OC) { ssum[t] = 0.f; ssq[t] = 0.f; }
    __syncthreads();

    const size_t gid = (size_t)blockIdx.x * blockDim.x + t;
    const int b  = (int)(gid >> 10);
    const int p  = (int)(gid & 1023);
    const int py = p >> 5;
    const int px = p & 31;

    float acc[OC];
    #pragma unroll
    for (int oc = 0; oc < OC; ++oc) acc[oc] = bias[oc];

    #pragma unroll
    for (int ic = 0; ic < IC; ++ic) {
        const size_t cbase = ((size_t)(b * IC + ic)) << 10;
        float sc = 1.f, sh = 0.f;
        if (BF16IN) { sc = inScale[ic]; sh = inShift[ic]; }
        #pragma unroll
        for (int ky = -1; ky <= 1; ++ky) {
            const int iy = py + ky;
            if (iy < 0 || iy > 31) continue;
            #pragma unroll
            for (int kx = -1; kx <= 1; ++kx) {
                const int ix = px + kx;
                if (ix < 0 || ix > 31) continue;
                float v;
                if (BF16IN) v = lrelu(bf2f(inB[cbase + (iy << 5) + ix]) * sc + sh);
                else        v = inF[cbase + (iy << 5) + ix];
                const int wofs = ic * 9 + (ky + 1) * 3 + (kx + 1);
                #pragma unroll
                for (int oc = 0; oc < OC; ++oc)
                    acc[oc] += sw[oc * IC * 9 + wofs] * v;
            }
        }
    }

    #pragma unroll
    for (int oc = 0; oc < OC; ++oc) {
        outp[(((size_t)(b * OC + oc)) << 10) + p] = f2bf(acc[oc]);
        atomicAdd(&ssum[oc], acc[oc]);
        atomicAdd(&ssq[oc], acc[oc] * acc[oc]);
    }
    __syncthreads();
    if (t < OC) { atomicAdd(&gsum[t], ssum[t]); atomicAdd(&gsq[t], ssq[t]); }
}

// ---------------------------------------------------------------------------
// bf16 WMMA GEMM: C[M,N] = A[M,K] * W[N,K]^T + bias.
// Block = 8 waves along M, one shared 16-row B strip staged into LDS via TDM.
// grid.x = N/16, grid.y = M/128, dynamic LDS = 16*KDIM*2 bytes.
// ---------------------------------------------------------------------------
template<int KDIM>
__global__ void gemm_bf16_wmma(const unsigned short* __restrict__ A,
                               const unsigned short* __restrict__ W,
                               const float* __restrict__ bias,
                               float* __restrict__ Cf,
                               unsigned short* __restrict__ Cbf,
                               int N) {
    extern __shared__ unsigned short Bs[];   // [16][KDIM] bf16, LDS offset 0
    const int lane = threadIdx.x & 31;
    const int wave = threadIdx.x >> 5;
    const int l16  = lane & 15;
    const int h    = lane >> 4;
    const int mr   = (blockIdx.y * 8 + wave) * 16;
    const int nb   = blockIdx.x * 16;

    // stage W rows [nb, nb+16) into LDS: one TDM descriptor from wave 0
#if HAVE_TDM
    if (threadIdx.x < 32) {
        tdm_load_2d_bf16(W + (size_t)nb * KDIM, 0u, (unsigned)KDIM, 16u, 16u);
    }
#else
    for (int i = threadIdx.x; i < 16 * KDIM; i += blockDim.x)
        Bs[i] = W[(size_t)nb * KDIM + i];
#endif
    __syncthreads();

    const unsigned short* arow = A + (size_t)(mr + l16) * KDIM;
    const unsigned short* brow = Bs + (size_t)l16 * KDIM;

    v8f c = {};
    #pragma unroll 4
    for (int k0 = 0; k0 < KDIM; k0 += 32) {
        FragAB a, b;
        a.q[0] = *reinterpret_cast<const Q16*>(arow + k0 + h * 8);
        a.q[1] = *reinterpret_cast<const Q16*>(arow + k0 + 16 + h * 8);
        b.q[0] = *reinterpret_cast<const Q16*>(brow + k0 + h * 8);
        b.q[1] = *reinterpret_cast<const Q16*>(brow + k0 + 16 + h * 8);
        c = __builtin_amdgcn_wmma_f32_16x16x32_bf16(false, a.v, false, b.v,
                                                    (short)0, c, false, false);
    }
    const int n = nb + l16;
    const float bs = bias ? bias[n] : 0.f;
    #pragma unroll
    for (int v = 0; v < 8; ++v) {
        const int m = mr + v + 8 * h;
        float val = c[v] + bs;
        if (Cf)  Cf[(size_t)m * N + n] = val;
        if (Cbf) Cbf[(size_t)m * N + n] = f2bf(val);
    }
}

// ---------------------------------------------------------------------------
// VQ argmin: 16-row strip per wave, codebook staged through LDS in two 64 KB
// TDM chunks (1024 rows each).  argmin of (||c||^2 - 2 z.c) with cross-lane
// min-reduce within each 16-lane half. Dynamic LDS = 65536 bytes.
// ---------------------------------------------------------------------------
__global__ void vq_argmin_kernel(const unsigned short* __restrict__ Z,
                                 const unsigned short* __restrict__ CB,
                                 const float* __restrict__ cnorm,
                                 int* __restrict__ idxOut) {
    extern __shared__ unsigned short CBs[];  // [1024][32] bf16 chunk, LDS off 0
    const int lane = threadIdx.x & 31;
    const int wave = threadIdx.x >> 5;
    const int l16  = lane & 15;
    const int h    = lane >> 4;
    const int mr   = (blockIdx.x * 8 + wave) * 16;

    FragAB a;
    const unsigned short* arow = Z + (size_t)(mr + l16) * 32;
    a.q[0] = *reinterpret_cast<const Q16*>(arow + h * 8);
    a.q[1] = *reinterpret_cast<const Q16*>(arow + 16 + h * 8);

    float best[8]; int bidx[8];
    #pragma unroll
    for (int v = 0; v < 8; ++v) { best[v] = 3.4e38f; bidx[v] = 0; }

    for (int chunk = 0; chunk < 2; ++chunk) {
        __syncthreads();   // all waves done with previous chunk
#if HAVE_TDM
        if (threadIdx.x < 32) {
            tdm_load_2d_bf16(CB + (size_t)chunk * 1024 * 32, 0u, 32u, 1024u, 1024u);
        }
#else
        for (int i = threadIdx.x; i < 1024 * 32; i += blockDim.x)
            CBs[i] = CB[(size_t)chunk * 1024 * 32 + i];
#endif
        __syncthreads();

        for (int nb = 0; nb < 1024; nb += 16) {
            FragAB bf;
            const unsigned short* bp = CBs + (size_t)(nb + l16) * 32;
            bf.q[0] = *reinterpret_cast<const Q16*>(bp + h * 8);
            bf.q[1] = *reinterpret_cast<const Q16*>(bp + 16 + h * 8);
            v8f c = {};
            c = __builtin_amdgcn_wmma_f32_16x16x32_bf16(false, a.v, false, bf.v,
                                                        (short)0, c, false, false);
            const int   id = chunk * 1024 + nb + l16;
            const float cn = cnorm[id];
            #pragma unroll
            for (int v = 0; v < 8; ++v) {
                const float val = cn - 2.f * c[v];
                if (val < best[v]) { best[v] = val; bidx[v] = id; }
            }
        }
    }
    #pragma unroll
    for (int v = 0; v < 8; ++v) {
        float bv = best[v]; int bi = bidx[v];
        #pragma unroll
        for (int off = 8; off > 0; off >>= 1) {   // stays within 16-lane half
            const float ov = __shfl_xor(bv, off, 32);
            const int   oi = __shfl_xor(bi, off, 32);
            if (ov < bv || (ov == bv && oi < bi)) { bv = ov; bi = oi; }
        }
        if (l16 == 0) idxOut[mr + v + 8 * h] = bi;
    }
}

// gather quantized rows (bf16 for GEMM2), accumulate sum((q - z)^2)
__global__ void quantize_loss_kernel(const int* __restrict__ idx,
                                     const float* __restrict__ cb,
                                     const float* __restrict__ zf,
                                     unsigned short* __restrict__ qbf,
                                     float* __restrict__ acc) {
    __shared__ float s;
    if (threadIdx.x == 0) s = 0.f;
    __syncthreads();
    const size_t t = (size_t)blockIdx.x * blockDim.x + threadIdx.x;
    const int row = (int)(t >> 5);
    const int e   = (int)(t & 31);
    const float q = cb[(size_t)idx[row] * 32 + e];
    const float d = q - zf[t];
    qbf[t] = f2bf(q);
    atomicAdd(&s, d * d);
    __syncthreads();
    if (threadIdx.x == 0) atomicAdd(acc, s);
}

// ---------------------------------------------------------------------------
// host-side orchestration
// ---------------------------------------------------------------------------
extern "C" void kernel_launch(void* const* d_in, const int* in_sizes, int n_in,
                              void* d_out, int out_size, void* d_ws, size_t ws_size,
                              hipStream_t stream) {
    (void)in_sizes; (void)n_in; (void)out_size; (void)ws_size;

    const float* x        = (const float*)d_in[0];
    const float* conv1_w  = (const float*)d_in[1];
    const float* conv1_b  = (const float*)d_in[2];
    const float* bn1_g    = (const float*)d_in[3];
    const float* bn1_b    = (const float*)d_in[4];
    const float* lin_w    = (const float*)d_in[5];
    const float* lin_b    = (const float*)d_in[6];
    const float* codebook = (const float*)d_in[7];
    const float* lin2_w   = (const float*)d_in[8];
    const float* lin2_b   = (const float*)d_in[9];
    const float* rc1_w    = (const float*)d_in[10];
    const float* rc1_b    = (const float*)d_in[11];
    const float* rbn1_g   = (const float*)d_in[12];
    const float* rbn1_b   = (const float*)d_in[13];
    const float* rc2_w    = (const float*)d_in[14];
    const float* rc2_b    = (const float*)d_in[15];
    const float* rbn2_g   = (const float*)d_in[16];
    const float* rbn2_b   = (const float*)d_in[17];
    const float* rc3_w    = (const float*)d_in[18];
    const float* rc3_b    = (const float*)d_in[19];
    const float* rbn3_g   = (const float*)d_in[20];
    const float* rbn3_b   = (const float*)d_in[21];
    const float* fconv_w  = (const float*)d_in[22];
    const float* fconv_b  = (const float*)d_in[23];
    const float* fbn_g    = (const float*)d_in[24];
    const float* fbn_b    = (const float*)d_in[25];

    float* out = (float*)d_out;

    // ---- workspace carve ----
    char* base = (char*)d_ws;
    size_t o = 0;
    auto alloc = [&](size_t bytes) -> char* {
        o = (o + 255) & ~(size_t)255;
        char* p = base + o; o += bytes; return p;
    };
    unsigned short* W1bf = (unsigned short*)alloc((size_t)512 * 2048 * 2);
    unsigned short* W2bf = (unsigned short*)alloc((size_t)2048 * 512 * 2);
    unsigned short* CBbf = (unsigned short*)alloc((size_t)2048 * 32 * 2);
    float* cnorm = (float*)alloc(2048 * 4);
    float* stats = (float*)alloc(512);
    float* ssum   = stats;        // [16]
    float* ssq    = stats + 16;   // [16]
    float* sscale = stats + 32;   // [16]
    float* sshift = stats + 48;   // [16]
    float* lossAcc = stats + 64;  // [1]
    float* y2 = (float*)alloc((size_t)4096 * 2048 * 4);        // f32 trunk
    char* big = alloc(218103808ULL);                            // phase-overlapped

    // phase A (encoder / VQ)
    unsigned short* t0  = (unsigned short*)(big);               // [4096,2048] bf16
    unsigned short* ybf = (unsigned short*)(big + 16777216);    // [4096,2048] bf16
    float*          zf  = (float*)         (big + 33554432);    // [4096,512]  f32
    unsigned short* zbf = (unsigned short*)(big + 41943040);    // [4096,512]  bf16
    unsigned short* qbf = (unsigned short*)(big + 46137344);    // [65536,32]  bf16
    int*            idx = (int*)           (big + 50331648);    // [65536]
    // phase R (refine) — aliases phase A
    unsigned short* r1   = (unsigned short*)(big);              // [4096,8,32,32]
    unsigned short* r2   = (unsigned short*)(big + 67108864);   // [4096,16,32,32]
    unsigned short* r3   = (unsigned short*)(big + 201326592);  // [4096,2,32,32]
    unsigned short* fraw = (unsigned short*)(big);              // [4096,2,32,32]

    const float invN = 1.0f / (4096.0f * 1024.0f);
    const int PIXB = 16384;   // (4096*1024)/256 conv blocks
    const int EWB  = 32768;   // 8388608/256 elementwise blocks

    // ---- weight conversions ----
    cvt_bf16_kernel<<<4096, 256, 0, stream>>>(lin_w,  W1bf, 512 * 2048);
    cvt_bf16_kernel<<<4096, 256, 0, stream>>>(lin2_w, W2bf, 2048 * 512);
    cvt_bf16_kernel<<<256, 256, 0, stream>>>(codebook, CBbf, 2048 * 32);
    cnorm_kernel<<<8, 256, 0, stream>>>(codebook, cnorm);

    // ---- encoder conv1 + BN + lrelu ----
    zero_f32<<<1, 32, 0, stream>>>(stats, 32);
    conv3x3_kernel<2, 2, false><<<PIXB, 256, 0, stream>>>(
        x, nullptr, nullptr, nullptr, conv1_w, conv1_b, t0, ssum, ssq);
    bn_finalize<<<1, 32, 0, stream>>>(ssum, ssq, bn1_g, bn1_b, sscale, sshift, 2, invN);
    bnlrelu_kernel<<<EWB, 256, 0, stream>>>(t0, sscale, sshift, ybf);

    // ---- GEMM1: z = y @ lin_w^T + b  (M=4096, N=512, K=2048), TDM B-staging ----
    gemm_bf16_wmma<2048><<<dim3(32, 32), 256, 16 * 2048 * 2, stream>>>(
        ybf, W1bf, lin_b, zf, zbf, 512);

    // ---- VQ argmin (TDM codebook staging) + quantize + loss ----
    vq_argmin_kernel<<<512, 256, 65536, stream>>>(zbf, CBbf, cnorm, idx);
    zero_f32<<<1, 32, 0, stream>>>(lossAcc, 1);
    quantize_loss_kernel<<<8192, 256, 0, stream>>>(idx, codebook, zf, qbf, lossAcc);
    loss_write_kernel<<<1, 1, 0, stream>>>(lossAcc, out);

    // ---- GEMM2: y2 = q @ lin2_w^T + b2  (M=4096, N=2048, K=512) ----
    gemm_bf16_wmma<512><<<dim3(128, 32), 256, 16 * 512 * 2, stream>>>(
        qbf, W2bf, lin2_b, y2, nullptr, 2048);

    // ---- refineNet x2 ----
    for (int it = 0; it < 2; ++it) {
        zero_f32<<<1, 32, 0, stream>>>(stats, 32);
        conv3x3_kernel<2, 8, false><<<PIXB, 256, 0, stream>>>(
            y2, nullptr, nullptr, nullptr, rc1_w, rc1_b, r1, ssum, ssq);
        bn_finalize<<<1, 32, 0, stream>>>(ssum, ssq, rbn1_g, rbn1_b, sscale, sshift, 8, invN);

        zero_f32<<<1, 32, 0, stream>>>(stats, 32);
        conv3x3_kernel<8, 16, true><<<PIXB, 256, 0, stream>>>(
            nullptr, r1, sscale, sshift, rc2_w, rc2_b, r2, ssum, ssq);
        bn_finalize<<<1, 32, 0, stream>>>(ssum, ssq, rbn2_g, rbn2_b, sscale, sshift, 16, invN);

        zero_f32<<<1, 32, 0, stream>>>(stats, 32);
        conv3x3_kernel<16, 2, true><<<PIXB, 256, 0, stream>>>(
            nullptr, r2, sscale, sshift, rc3_w, rc3_b, r3, ssum, ssq);
        bn_finalize<<<1, 32, 0, stream>>>(ssum, ssq, rbn3_g, rbn3_b, sscale, sshift, 2, invN);

        residual_kernel<<<EWB, 256, 0, stream>>>(y2, r3, sscale, sshift);
    }

    // ---- final conv + BN + sigmoid ----
    zero_f32<<<1, 32, 0, stream>>>(stats, 32);
    conv3x3_kernel<2, 2, false><<<PIXB, 256, 0, stream>>>(
        y2, nullptr, nullptr, nullptr, fconv_w, fconv_b, fraw, ssum, ssq);
    bn_finalize<<<1, 32, 0, stream>>>(ssum, ssq, fbn_g, fbn_b, sscale, sshift, 2, invN);
    sigmoid_kernel<<<EWB, 256, 0, stream>>>(fraw, sscale, sshift, out + 1);
}
